// ColbertLoss_27822798143930
// MI455X (gfx1250) — matrix operand order repaired and checked by reading, not compiled, not run on hardware
//
#include <hip/hip_runtime.h>

// ---------------------------------------------------------------------------
// ColBERT MaxSim loss for MI455X (gfx1250, wave32, WMMA)
//   B=64 batches, NQ=32 query tokens, ND=1024 doc tokens, D=128 dim
//   Kernel 1: f16 WMMA GEMM + running max, double-buffered LDS pipeline
//   Kernel 2: log-softmax loss on the 64x64 score matrix
// ---------------------------------------------------------------------------

typedef _Float16 v16h __attribute__((ext_vector_type(16)));
typedef _Float16 v8h  __attribute__((ext_vector_type(8)));
typedef _Float16 v4h  __attribute__((ext_vector_type(4)));
typedef float    v8f  __attribute__((ext_vector_type(8)));

#define B_      64
#define NQ_     32
#define ND_     1024
#define D_      128
#define SCHUNK  128                  // doc rows staged in LDS per iteration
#define NCHUNK  (ND_ / SCHUNK)       // 8
#define NREG    (SCHUNK * D_ / 4 / 256)  // float4s per thread per chunk = 16
#define TEMP    0.02f

__device__ __forceinline__ v8f v8f_max(v8f a, v8f b) {
  v8f r;
#pragma unroll
  for (int i = 0; i < 8; ++i) r[i] = fmaxf(a[i], b[i]);
  return r;
}

__device__ __forceinline__ v8h cvt8(const float4 a, const float4 b) {
  v8h r;
  r[0] = (_Float16)a.x; r[1] = (_Float16)a.y;
  r[2] = (_Float16)a.z; r[3] = (_Float16)a.w;
  r[4] = (_Float16)b.x; r[5] = (_Float16)b.y;
  r[6] = (_Float16)b.z; r[7] = (_Float16)b.w;
  return r;
}

// ---------------------------------------------------------------------------
// Kernel 1: scores[b,c] = sum_n max_s <q[b,n,:], d[c,s,:]>
// grid = (c, b_group) = (64, 8); block = 256 threads = 8 waves; wave w -> b
// ---------------------------------------------------------------------------
__global__ __launch_bounds__(256)
void colbert_scores_kernel(const float* __restrict__ q,
                           const float* __restrict__ d,
                           float* __restrict__ scores) {
  __shared__ _Float16 d_lds[2][SCHUNK * D_];       // 2 x 32 KB f16 doc chunks

  const int c    = blockIdx.x;                     // doc batch
  const int bg   = blockIdx.y;                     // group of 8 query batches
  const int tid  = threadIdx.x;
  const int wave = tid >> 5;
  const int lane = tid & 31;
  const int b    = bg * 8 + wave;

  const int m  = lane & 15;                        // row (A) / col (B) in tile
  const int hi = lane >> 4;                        // half-wave select

  // ---- stage this wave's q[b] as register-resident WMMA A fragments ----
  // ISA 16-bit A 16x32 layout: lane(M=lane&15), half e -> K = (e&8)*2 + hi*8 + (e&7)
  v16h afrag[2][4];
#pragma unroll
  for (int qt = 0; qt < 2; ++qt) {
#pragma unroll
    for (int kc = 0; kc < 4; ++kc) {
      const float* p =
          q + ((size_t)(b * NQ_ + qt * 16 + m) * D_ + kc * 32 + hi * 8);
      float4 f0 = *(const float4*)(p);             // K = kc*32 + hi*8 + 0..3
      float4 f1 = *(const float4*)(p + 4);         //                 + 4..7
      float4 f2 = *(const float4*)(p + 16);        // K = +16 + hi*8 + 0..3
      float4 f3 = *(const float4*)(p + 20);        //                 + 4..7
      v8h lo = cvt8(f0, f1);
      v8h hp = cvt8(f2, f3);
      afrag[qt][kc] = __builtin_shufflevector(
          lo, hp, 0, 1, 2, 3, 4, 5, 6, 7, 8, 9, 10, 11, 12, 13, 14, 15);
    }
  }

  v8f vmax0, vmax1;
#pragma unroll
  for (int i = 0; i < 8; ++i) { vmax0[i] = -1e30f; vmax1[i] = -1e30f; }

  const float* dc = d + (size_t)c * ND_ * D_;
  float4 regs[NREG];

  // issue global loads for chunk `s0` into registers (coalesced float4s)
  auto load_chunk = [&](int s0) {
#pragma unroll
    for (int j = 0; j < NREG; ++j) {
      const int i = tid + j * 256;                 // float4 index in chunk
      regs[j] = *(const float4*)(dc + (size_t)s0 * D_ + (size_t)i * 4);
    }
  };
  // convert f32 -> f16 and store into LDS buffer `buf`
  auto store_chunk = [&](int buf) {
#pragma unroll
    for (int j = 0; j < NREG; ++j) {
      const int i = tid + j * 256;
      v4h h;
      h[0] = (_Float16)regs[j].x; h[1] = (_Float16)regs[j].y;
      h[2] = (_Float16)regs[j].z; h[3] = (_Float16)regs[j].w;
      *(v4h*)&d_lds[buf][i * 4] = h;
    }
  };

  // ---- software-pipelined: prefetch chunk i+1 while WMMAs run on chunk i ----
  load_chunk(0);
  store_chunk(0);
  __syncthreads();

  for (int ci = 0; ci < NCHUNK; ++ci) {
    if (ci + 1 < NCHUNK) load_chunk((ci + 1) * SCHUNK);   // overlaps compute

    const _Float16* cur = d_lds[ci & 1];
#pragma unroll
    for (int st = 0; st < SCHUNK / 16; ++st) {
      v8f acc0 = {};
      v8f acc1 = {};
#pragma unroll
      for (int kc = 0; kc < 4; ++kc) {
        // B 32x16 f16 layout: lane = N (col) per half, K = (e&8)*2 + hi*8 + (e&7)
        // column-major B == row-major d, so these are two 16B LDS reads
        const _Float16* bp = &cur[(st * 16 + m) * D_ + kc * 32 + hi * 8];
        v8h blo = *(const v8h*)(bp);
        v8h bhi = *(const v8h*)(bp + 16);
        v16h bfrag = __builtin_shufflevector(
            blo, bhi, 0, 1, 2, 3, 4, 5, 6, 7, 8, 9, 10, 11, 12, 13, 14, 15);
        acc0 = __builtin_amdgcn_wmma_f32_16x16x32_f16(
            false, afrag[0][kc], false, bfrag, (short)0, acc0, false, false);
        acc1 = __builtin_amdgcn_wmma_f32_16x16x32_f16(
            false, afrag[1][kc], false, bfrag, (short)0, acc1, false, false);
      }
      vmax0 = v8f_max(vmax0, acc0);                // running max over s-tiles
      vmax1 = v8f_max(vmax1, acc1);
    }

    if (ci + 1 < NCHUNK) store_chunk((ci + 1) & 1);       // fill other buffer
    __syncthreads();
  }

  // ---- MaxSim reduction ----
  // C/D layout: VGPR r holds row M=r (lanes 0-15) and M=r+8 (lanes 16-31),
  // with N = lane&15.  Max across the 16-lane half = max over doc columns.
  float rowsum = 0.0f;
#pragma unroll
  for (int r = 0; r < 8; ++r) {
    float v0 = vmax0[r];
    float v1 = vmax1[r];
#pragma unroll
    for (int off = 1; off < 16; off <<= 1) {
      v0 = fmaxf(v0, __shfl_xor(v0, off, 32));
      v1 = fmaxf(v1, __shfl_xor(v1, off, 32));
    }
    rowsum += v0 + v1;                             // sum over query rows
  }
  rowsum += __shfl_xor(rowsum, 16, 32);            // combine the two halves

  if (lane == 0) scores[b * B_ + c] = rowsum;
}

// ---------------------------------------------------------------------------
// Kernel 2: loss = -mean_b log_softmax(scores[b,:]/q_len[b]/TEMP)[b]
// ---------------------------------------------------------------------------
__global__ __launch_bounds__(B_)
void colbert_loss_kernel(const float* __restrict__ q,
                         const float* __restrict__ scores,
                         float* __restrict__ out) {
  __shared__ float red[B_];
  const int b = threadIdx.x;

  int qlen = 0;
#pragma unroll
  for (int n = 0; n < NQ_; ++n)
    qlen += (q[(size_t)(b * NQ_ + n) * D_] != 0.0f) ? 1 : 0;
  if (qlen == 0) qlen = 1;
  const float scale = 1.0f / ((float)qlen * TEMP);

  float mx = -1e30f;
  for (int cc = 0; cc < B_; ++cc)
    mx = fmaxf(mx, scores[b * B_ + cc] * scale);
  float sum = 0.0f;
  for (int cc = 0; cc < B_; ++cc)
    sum += __expf(scores[b * B_ + cc] * scale - mx);
  const float logp = scores[b * B_ + b] * scale - mx - __logf(sum);

  red[b] = -logp;
  __syncthreads();
  if (b == 0) {
    float acc = 0.0f;
    for (int i = 0; i < B_; ++i) acc += red[i];
    out[0] = acc / (float)B_;
  }
}

// ---------------------------------------------------------------------------
extern "C" void kernel_launch(void* const* d_in, const int* in_sizes, int n_in,
                              void* d_out, int out_size, void* d_ws,
                              size_t ws_size, hipStream_t stream) {
  (void)in_sizes; (void)n_in; (void)out_size; (void)ws_size;
  const float* q = (const float*)d_in[0];   // (64, 32, 128) f32
  const float* d = (const float*)d_in[1];   // (64, 1024, 128) f32
  float* scores  = (float*)d_ws;            // 64*64 f32 scratch

  colbert_scores_kernel<<<dim3(B_, B_ / 8), 256, 0, stream>>>(q, d, scores);
  colbert_loss_kernel<<<1, B_, 0, stream>>>(q, scores, (float*)d_out);
}
